// RelativeMultiHeadedSelfAttentionModule_42992622633325
// MI455X (gfx1250) — matrix operand order repaired and checked by reading, not compiled, not run on hardware
//
#include <hip/hip_runtime.h>
#include <hip/hip_bf16.h>
#include <math.h>

#define CDIM 512
#define HEADS 8
#define HD 64
#define BATCH 8
#define SEQ 1024
#define PB 2              // batches processed per attention pass

typedef _Float16 h16;
typedef __attribute__((ext_vector_type(8)))  _Float16 v8h;
typedef __attribute__((ext_vector_type(16))) _Float16 v16h;
typedef __attribute__((ext_vector_type(8)))  float    v8f;

#define WMMA_F16(a, b, c) \
  __builtin_amdgcn_wmma_f32_16x16x32_f16(false, (a), false, (b), (short)0, (c), false, false)

// ---- WMMA fragment loaders (ISA 7.12.2 layouts) --------------------------
// A 16x32 f16: lane holds row M=lane%16; K elements: base=(lane/16)*8,
//   e<8 -> K=base+e ; e>=8 -> K=16+base+(e-8). Two contiguous 16B chunks.
__device__ __forceinline__ v16h load_fragA(const h16* __restrict__ base, int ld,
                                           int row, int k0, int lane) {
  const h16* p = base + (size_t)(row + (lane & 15)) * ld + (k0 + ((lane >> 4) << 3));
  v8h lo = *(const v8h*)p;
  v8h hi = *(const v8h*)(p + 16);
  v16h r;
#pragma unroll
  for (int i = 0; i < 8; ++i) { r[i] = lo[i]; r[i + 8] = hi[i]; }
  return r;
}

// B 32x16 f16 (given as Bt row-major [N,K]): lane holds col n=lane%16;
//   K elements: base=(lane/16)*8, e<8 -> K=k0+base+e ; e>=8 -> K=k0+16+base+(e-8).
__device__ __forceinline__ v16h load_fragB(const h16* __restrict__ base, int ld,
                                           int col, int k0, int lane) {
  const h16* p = base + (size_t)(col + (lane & 15)) * ld + (k0 + ((lane >> 4) << 3));
  v8h lo = *(const v8h*)p;
  v8h hi = *(const v8h*)(p + 16);
  v16h r;
#pragma unroll
  for (int i = 0; i < 8; ++i) { r[i] = lo[i]; r[i + 8] = hi[i]; }
  return r;
}

// ---- 32x64 wave tile GEMM core: 2 A-frags, 8 accumulators ----------------
template <int KDIM, bool PREFETCH>
__device__ __forceinline__ void gemm_core_32x64(const h16* __restrict__ A, int ldA,
                                                const h16* __restrict__ Bt, int ldB,
                                                int rowBase, int colBase, int lane,
                                                v8f acc[2][4]) {
  for (int k0 = 0; k0 < KDIM; k0 += 32) {
    if (PREFETCH && k0 + 32 < KDIM) {
      __builtin_prefetch(A + (size_t)(rowBase + (lane & 15)) * ldA + k0 + 32, 0, 1);
      __builtin_prefetch(Bt + (size_t)(colBase + (lane & 15)) * ldB + k0 + 32, 0, 1);
    }
    v16h a0 = load_fragA(A, ldA, rowBase, k0, lane);
    v16h a1 = load_fragA(A, ldA, rowBase + 16, k0, lane);
#pragma unroll
    for (int t = 0; t < 4; ++t) {
      v16h b = load_fragB(Bt, ldB, colBase + t * 16, k0, lane);
      acc[0][t] = WMMA_F16(a0, b, acc[0][t]);
      acc[1][t] = WMMA_F16(a1, b, acc[1][t]);
    }
  }
}

// ---- prep kernels --------------------------------------------------------
__global__ void k_pe(h16* __restrict__ pe) {
  int idx = blockIdx.x * blockDim.x + threadIdx.x;
  if (idx >= SEQ * CDIM) return;
  int pos = idx >> 9, c = idx & (CDIM - 1);
  float div = __expf(-(float)(c >> 1) * (9.210340371976184f / 256.0f));
  float ang = (float)pos * div;
  pe[idx] = (h16)((c & 1) ? cosf(ang) : sinf(ang));
}

__global__ void k_cvt(const float* __restrict__ in, h16* __restrict__ out, int n) {
  int idx = blockIdx.x * blockDim.x + threadIdx.x;
  if (idx < n) out[idx] = (h16)in[idx];
}

// W [K, Nc] row-major fp32 -> Wt [Nc, K] row-major f16
__global__ void k_transpose(const float* __restrict__ W, h16* __restrict__ Wt,
                            int K, int Nc) {
  int idx = blockIdx.x * blockDim.x + threadIdx.x;
  if (idx >= K * Nc) return;
  int n = idx / K, k = idx - n * K;
  Wt[idx] = (h16)W[(size_t)k * Nc + n];
}

// ---- GEMM kernels: wave = 32 rows x 64 cols, block = 4 waves (128x64) ----
__global__ __launch_bounds__(128) void k_gemm_qkv(
    const h16* __restrict__ A, const h16* __restrict__ Bt,
    const float* __restrict__ ub, const float* __restrict__ vb,
    h16* __restrict__ qu, h16* __restrict__ qv,
    h16* __restrict__ kh, h16* __restrict__ vt) {
  int lane = threadIdx.x & 31, wave = threadIdx.x >> 5;
  int rowBase = (blockIdx.y * 4 + wave) * 32;
  int colBase = blockIdx.x * 64;
  v8f z = {};
  v8f acc[2][4] = {{z, z, z, z}, {z, z, z, z}};
  gemm_core_32x64<CDIM, true>(A, CDIM, Bt, CDIM, rowBase, colBase, lane, acc);
  int rOff = (lane >> 4) << 3;
#pragma unroll
  for (int i = 0; i < 2; ++i)
#pragma unroll
    for (int t = 0; t < 4; ++t)
#pragma unroll
      for (int r = 0; r < 8; ++r) {
        int row = rowBase + i * 16 + r + rOff;
        int col = colBase + t * 16 + (lane & 15);
        int b = row >> 10, n = row & (SEQ - 1);
        int s = col >> 9, hd = col & 511, h = hd >> 6, d = hd & 63;
        float val = acc[i][t][r];
        size_t qidx = (((size_t)(b * HEADS + h) * SEQ) + n) * HD + d;
        if (s == 0) {
          qu[qidx] = (h16)(val + ub[hd]);
          qv[qidx] = (h16)(val + vb[hd]);
        } else if (s == 1) {
          kh[qidx] = (h16)val;
        } else {
          vt[(((size_t)(b * HEADS + h) * HD) + d) * SEQ + n] = (h16)val;
        }
      }
}

__global__ __launch_bounds__(128) void k_gemm_p(
    const h16* __restrict__ A, const h16* __restrict__ Bt, h16* __restrict__ ph) {
  int lane = threadIdx.x & 31, wave = threadIdx.x >> 5;
  int rowBase = (blockIdx.y * 4 + wave) * 32;
  int colBase = blockIdx.x * 64;
  v8f z = {};
  v8f acc[2][4] = {{z, z, z, z}, {z, z, z, z}};
  gemm_core_32x64<CDIM, true>(A, CDIM, Bt, CDIM, rowBase, colBase, lane, acc);
  int rOff = (lane >> 4) << 3;
#pragma unroll
  for (int i = 0; i < 2; ++i)
#pragma unroll
    for (int t = 0; t < 4; ++t)
#pragma unroll
      for (int r = 0; r < 8; ++r) {
        int m = rowBase + i * 16 + r + rOff;
        int col = colBase + t * 16 + (lane & 15);
        int h = col >> 6, d = col & 63;
        ph[((size_t)h * SEQ + m) * HD + d] = (h16)acc[i][t][r];
      }
}

// P[z,n,m] = (q+v_bias)[b0+c,h,n,:] . p[h,m,:]   z = c*HEADS + h
__global__ __launch_bounds__(128) void k_gemm_pos(
    const h16* __restrict__ qv, const h16* __restrict__ ph,
    h16* __restrict__ P, int b0) {
  int zi = blockIdx.z;
  int b = b0 + zi / HEADS, h = zi % HEADS;
  const h16* A  = qv + (size_t)(b * HEADS + h) * SEQ * HD;
  const h16* Bt = ph + (size_t)h * SEQ * HD;
  int lane = threadIdx.x & 31, wave = threadIdx.x >> 5;
  int rowBase = (blockIdx.y * 4 + wave) * 32;
  int colBase = blockIdx.x * 64;
  v8f z = {};
  v8f acc[2][4] = {{z, z, z, z}, {z, z, z, z}};
  gemm_core_32x64<HD, false>(A, HD, Bt, HD, rowBase, colBase, lane, acc);
  int rOff = (lane >> 4) << 3;
#pragma unroll
  for (int i = 0; i < 2; ++i)
#pragma unroll
    for (int t = 0; t < 4; ++t)
#pragma unroll
      for (int r = 0; r < 8; ++r) {
        int n = rowBase + i * 16 + r + rOff;
        int m = colBase + t * 16 + (lane & 15);
        P[((size_t)zi * SEQ + n) * SEQ + m] = (h16)acc[i][t][r];
      }
}

// logits[z,n,m] = (content + rel_shift(P)) * scale
__global__ __launch_bounds__(128) void k_gemm_content(
    const h16* __restrict__ qu, const h16* __restrict__ kh,
    const h16* __restrict__ P, float* __restrict__ Lb, int b0) {
  int zi = blockIdx.z;
  int b = b0 + zi / HEADS, h = zi % HEADS;
  const h16* A  = qu + (size_t)(b * HEADS + h) * SEQ * HD;
  const h16* Bt = kh + (size_t)(b * HEADS + h) * SEQ * HD;
  int lane = threadIdx.x & 31, wave = threadIdx.x >> 5;
  int rowBase = (blockIdx.y * 4 + wave) * 32;
  int colBase = blockIdx.x * 64;
  v8f z = {};
  v8f acc[2][4] = {{z, z, z, z}, {z, z, z, z}};
  gemm_core_32x64<HD, false>(A, HD, Bt, HD, rowBase, colBase, lane, acc);
  const h16* Ph = P + (size_t)zi * SEQ * SEQ;
  int rOff = (lane >> 4) << 3;
#pragma unroll
  for (int i = 0; i < 2; ++i)
#pragma unroll
    for (int t = 0; t < 4; ++t)
#pragma unroll
      for (int r = 0; r < 8; ++r) {
        int n = rowBase + i * 16 + r + rOff;
        int m = colBase + t * 16 + (lane & 15);
        float ps;
        if (m <= n)          ps = (float)Ph[(size_t)n * SEQ + (SEQ - 1) + m - n];
        else if (m == n + 1) ps = 0.0f;
        else                 ps = (float)Ph[(size_t)(n + 1) * SEQ + (m - n - 2)];
        Lb[((size_t)zi * SEQ + n) * SEQ + m] = (acc[i][t][r] + ps) * 0.125f;
      }
}

__global__ __launch_bounds__(128) void k_softmax(
    const float* __restrict__ Lb, h16* __restrict__ Ah) {
  int n = blockIdx.x, zi = blockIdx.y;
  const float* row = Lb + ((size_t)zi * SEQ + n) * SEQ;
  h16* orow = Ah + ((size_t)zi * SEQ + n) * SEQ;
  int tid = threadIdx.x, lane = tid & 31, wave = tid >> 5;
  float v[8], mx = -3.0e38f;
#pragma unroll
  for (int i = 0; i < 8; ++i) { v[i] = row[tid + i * 128]; mx = fmaxf(mx, v[i]); }
#pragma unroll
  for (int off = 16; off > 0; off >>= 1) mx = fmaxf(mx, __shfl_xor(mx, off, 32));
  __shared__ float red[4];
  if (lane == 0) red[wave] = mx;
  __syncthreads();
  mx = fmaxf(fmaxf(red[0], red[1]), fmaxf(red[2], red[3]));
  __syncthreads();
  float sum = 0.0f;
#pragma unroll
  for (int i = 0; i < 8; ++i) { v[i] = __expf(v[i] - mx); sum += v[i]; }
#pragma unroll
  for (int off = 16; off > 0; off >>= 1) sum += __shfl_xor(sum, off, 32);
  if (lane == 0) red[wave] = sum;
  __syncthreads();
  float inv = 1.0f / (red[0] + red[1] + red[2] + red[3]);
#pragma unroll
  for (int i = 0; i < 8; ++i) orow[tid + i * 128] = (h16)(v[i] * inv);
}

// ctx[b,n,h*64+d] = sum_m attn[z,n,m] * v[b,h,m,d]
__global__ __launch_bounds__(128) void k_gemm_av(
    const h16* __restrict__ Ah, const h16* __restrict__ vt,
    h16* __restrict__ ctx, int b0) {
  int zi = blockIdx.z;
  int b = b0 + zi / HEADS, h = zi % HEADS;
  const h16* A  = Ah + (size_t)zi * SEQ * SEQ;              // [N, N]
  const h16* Bt = vt + (size_t)(b * HEADS + h) * HD * SEQ;  // [64, N]
  int lane = threadIdx.x & 31, wave = threadIdx.x >> 5;
  int rowBase = (blockIdx.y * 4 + wave) * 32;
  v8f z = {};
  v8f acc[2][4] = {{z, z, z, z}, {z, z, z, z}};
  gemm_core_32x64<SEQ, true>(A, SEQ, Bt, SEQ, rowBase, 0, lane, acc);
  int rOff = (lane >> 4) << 3;
#pragma unroll
  for (int i = 0; i < 2; ++i)
#pragma unroll
    for (int t = 0; t < 4; ++t)
#pragma unroll
      for (int r = 0; r < 8; ++r) {
        int n = rowBase + i * 16 + r + rOff;
        int d = t * 16 + (lane & 15);
        ctx[((size_t)(b * SEQ + n)) * CDIM + h * HD + d] = (h16)acc[i][t][r];
      }
}

__global__ __launch_bounds__(128) void k_gemm_out(
    const h16* __restrict__ A, const h16* __restrict__ Bt,
    const float* __restrict__ bproj, float* __restrict__ out) {
  int lane = threadIdx.x & 31, wave = threadIdx.x >> 5;
  int rowBase = (blockIdx.y * 4 + wave) * 32;
  int colBase = blockIdx.x * 64;
  v8f z = {};
  v8f acc[2][4] = {{z, z, z, z}, {z, z, z, z}};
  gemm_core_32x64<CDIM, true>(A, CDIM, Bt, CDIM, rowBase, colBase, lane, acc);
  int rOff = (lane >> 4) << 3;
#pragma unroll
  for (int i = 0; i < 2; ++i)
#pragma unroll
    for (int t = 0; t < 4; ++t)
#pragma unroll
      for (int r = 0; r < 8; ++r) {
        int row = rowBase + i * 16 + r + rOff;
        int col = colBase + t * 16 + (lane & 15);
        out[(size_t)row * CDIM + col] = acc[i][t][r] + bproj[col];
      }
}

// ---- host-side launcher --------------------------------------------------
extern "C" void kernel_launch(void* const* d_in, const int* in_sizes, int n_in,
                              void* d_out, int out_size, void* d_ws, size_t ws_size,
                              hipStream_t stream) {
  const float* x     = (const float*)d_in[0];
  const float* Wqkv  = (const float*)d_in[1];
  const float* Wpos  = (const float*)d_in[2];
  const float* ub    = (const float*)d_in[3];
  const float* vb    = (const float*)d_in[4];
  const float* Wproj = (const float*)d_in[5];
  const float* bproj = (const float*)d_in[6];
  float* out = (float*)d_out;

  char* ws = (char*)d_ws;
  size_t off = 0;
  auto alloc = [&](size_t bytes) -> void* {
    void* p = ws + off;
    off += (bytes + 255) & ~(size_t)255;
    return p;
  };

  const size_t BND = (size_t)BATCH * SEQ * CDIM;       // 4M elems
  const size_t QSZ = (size_t)BATCH * HEADS * SEQ * HD; // 4M elems
  const size_t SCORES = (size_t)PB * HEADS * SEQ * SEQ;

  h16*  x_h    = (h16*)alloc(BND * 2);
  h16*  wqkvT  = (h16*)alloc((size_t)3 * CDIM * CDIM * 2);
  h16*  wposT  = (h16*)alloc((size_t)CDIM * CDIM * 2);
  h16*  wprojT = (h16*)alloc((size_t)CDIM * CDIM * 2);
  h16*  pe_h   = (h16*)alloc((size_t)SEQ * CDIM * 2);
  h16*  p_h    = (h16*)alloc((size_t)HEADS * SEQ * HD * 2);
  h16*  qu     = (h16*)alloc(QSZ * 2);
  h16*  qv     = (h16*)alloc(QSZ * 2);
  h16*  kh     = (h16*)alloc(QSZ * 2);
  h16*  vt     = (h16*)alloc(QSZ * 2);
  h16*  ctx    = (h16*)alloc(BND * 2);
  h16*  P      = (h16*)alloc(SCORES * 2);
  float* Lb    = (float*)alloc(SCORES * 4);
  h16*  Ah     = (h16*)alloc(SCORES * 2);

  // prep
  k_pe<<<(SEQ * CDIM + 255) / 256, 256, 0, stream>>>(pe_h);
  k_cvt<<<((int)BND + 255) / 256, 256, 0, stream>>>(x, x_h, (int)BND);
  k_transpose<<<(3 * CDIM * CDIM + 255) / 256, 256, 0, stream>>>(Wqkv, wqkvT, CDIM, 3 * CDIM);
  k_transpose<<<(CDIM * CDIM + 255) / 256, 256, 0, stream>>>(Wpos, wposT, CDIM, CDIM);
  k_transpose<<<(CDIM * CDIM + 255) / 256, 256, 0, stream>>>(Wproj, wprojT, CDIM, CDIM);

  // projections
  k_gemm_qkv<<<dim3(3 * CDIM / 64, BATCH * SEQ / 128), 128, 0, stream>>>(
      x_h, wqkvT, ub, vb, qu, qv, kh, vt);
  k_gemm_p<<<dim3(CDIM / 64, SEQ / 128), 128, 0, stream>>>(pe_h, wposT, p_h);

  // attention, PB batches per pass to bound workspace
  for (int b0 = 0; b0 < BATCH; b0 += PB) {
    k_gemm_pos<<<dim3(SEQ / 64, SEQ / 128, PB * HEADS), 128, 0, stream>>>(qv, p_h, P, b0);
    k_gemm_content<<<dim3(SEQ / 64, SEQ / 128, PB * HEADS), 128, 0, stream>>>(qu, kh, P, Lb, b0);
    k_softmax<<<dim3(SEQ, PB * HEADS), 128, 0, stream>>>(Lb, Ah);
    k_gemm_av<<<dim3(1, SEQ / 128, PB * HEADS), 128, 0, stream>>>(Ah, vt, ctx, b0);
  }

  // output projection
  k_gemm_out<<<dim3(CDIM / 64, BATCH * SEQ / 128), 128, 0, stream>>>(
      ctx, wprojT, bproj, out);

  (void)in_sizes; (void)n_in; (void)out_size; (void)ws_size;
}
